// my_embedding_5978594476294
// MI455X (gfx1250) — compile-verified, matrix operand order checked by reading, not run on hardware
//
#include <hip/hip_runtime.h>
#include <stdint.h>

// ---------------- types ----------------
typedef __attribute__((ext_vector_type(16))) __bf16        v16bf;
typedef __attribute__((ext_vector_type(8)))  float         v8f;
typedef __attribute__((ext_vector_type(4)))  unsigned int  v4u;
typedef __attribute__((ext_vector_type(2)))  float         v2f;

typedef __attribute__((address_space(1))) int  gint;   // global (device) int
typedef __attribute__((address_space(3))) int  sint;   // LDS int

#define BM     128      // M rows per block
#define BN     64       // N cols per block
#define KREAL  300      // true K
#define KP     320      // K padded to 10 * 32
#define KL     328      // LDS row stride in bf16 elems (16B-aligned rows, conflict spread)
#define NTOTAL 2048
#define MTOTAL 32768

union FragU { v4u q[2]; v16bf v; };

__device__ __forceinline__ uint32_t pack_bf16x2(float a, float b) {
  uint32_t ua = __builtin_bit_cast(uint32_t, a);
  uint32_t ub = __builtin_bit_cast(uint32_t, b);
  ua = (ua + 0x7FFFu + ((ua >> 16) & 1u)) >> 16;   // round-to-nearest-even
  ub = (ub + 0x7FFFu + ((ub >> 16) & 1u)) >> 16;
  return ua | (ub << 16);
}

__global__ __launch_bounds__(256) void emb_linear_wmma(
    const int* __restrict__ x, const float* __restrict__ emb,
    const float* __restrict__ W, const float* __restrict__ bias,
    float* __restrict__ out)
{
  extern __shared__ __align__(16) char smem[];
  __bf16* As   = (__bf16*)smem;                                  // [BM][KL]
  __bf16* Bs   = (__bf16*)(smem + (size_t)BM * KL * 2);          // [BN][KL] (n-major, transposed W)
  int*    toks = (int*)   (smem + (size_t)(BM + BN) * KL * 2);   // [BM]

  const int tid  = threadIdx.x;
  const int lane = tid & 31;
  const int wave = tid >> 5;
  const int m0   = blockIdx.y * BM;
  const int n0   = blockIdx.x * BN;

  // ---- Phase 0: stage token IDs into LDS (async-tensor path on gfx1250) ----
#if __has_builtin(__builtin_amdgcn_global_load_async_to_lds_b32)
  if (tid < BM) {
    __builtin_amdgcn_global_load_async_to_lds_b32(
        (gint*)(x + m0 + tid),
        (sint*)(toks + tid),
        0, 0);
  }
#if __has_builtin(__builtin_amdgcn_s_wait_asynccnt)
  __builtin_amdgcn_s_wait_asynccnt(0);
#else
  asm volatile("s_wait_asynccnt 0" ::: "memory");
#endif
#else
  if (tid < BM) toks[tid] = x[m0 + tid];
#endif
  __syncthreads();

  // ---- Phase 1: gather embedding rows -> bf16 LDS A tile ----
  // wave w handles rows [16w, 16w+16); each row = 300 f32 (1200 B)
  {
    const int rbase = wave * 16;
    for (int rr = 0; rr < 16; ++rr) {
      const int r = rbase + rr;
      const long long tok = (long long)toks[r];
      const float* src = emb + tok * KREAL;
      if (r + 1 < BM) {  // prefetch next gathered row (global_prefetch_b8)
        const float* nxt = emb + (long long)toks[r + 1] * KREAL;
        __builtin_prefetch(nxt + lane * 4, 0, 0);
      }
#pragma unroll
      for (int p = 0; p < 4; ++p) {
        const int k = p * 64 + lane * 2;
        v2f v = *(const v2f*)(src + k);
        *(uint32_t*)(As + r * KL + k) = pack_bf16x2(v.x, v.y);
      }
      {
        const int k = 256 + lane * 2;        // tail: k = 256..318
        uint32_t pk = 0;
        int kd;
        if (k < KREAL) {
          v2f v = *(const v2f*)(src + k);
          pk = pack_bf16x2(v.x, v.y);
          kd = k;
        } else {
          kd = KREAL + (lane - 22) * 2;      // zero-fill k = 300..319
        }
        *(uint32_t*)(As + r * KL + kd) = pk;
      }
    }
  }

  // ---- Phase 2: W tile [KREAL][BN] -> transposed bf16 LDS B tile [BN][KL] ----
  for (int k = wave; k < KP; k += 8) {
    const int n = lane * 2;                  // covers n = 0..63
    if (k < KREAL) {
      v2f v = *(const v2f*)(W + (long long)k * NTOTAL + n0 + n);
      uint32_t pk = pack_bf16x2(v.x, v.y);
      *(uint16_t*)(Bs + (n + 0) * KL + k) = (uint16_t)pk;
      *(uint16_t*)(Bs + (n + 1) * KL + k) = (uint16_t)(pk >> 16);
    } else {                                 // zero-fill k = 300..319
      *(uint16_t*)(Bs + (n + 0) * KL + k) = 0;
      *(uint16_t*)(Bs + (n + 1) * KL + k) = 0;
    }
  }

  __syncthreads();

  // ---- Phase 3: WMMA main loop. Wave computes 32M x 32N (2x2 16x16 tiles) ----
  const int wm   = wave & 3;        // M: 4 waves * 32 rows
  const int wn   = wave >> 2;       // N: 2 waves * 32 cols
  const int half = lane >> 4;       // ISA 16-bit operand layout: half-wave K split
  const int l15  = lane & 15;

  v8f acc[2][2] = {};

#pragma unroll
  for (int ks = 0; ks < KP / 32; ++ks) {
    const int kb = ks * 32 + half * 8;       // per-lane K base (bf16 elems)

    v16bf aF[2], bF[2];
#pragma unroll
    for (int mi = 0; mi < 2; ++mi) {         // A: rows m, K 0..7 and 16..23 (+half*8)
      const __bf16* ap = As + (wm * 32 + mi * 16 + l15) * KL + kb;
      FragU f;
      f.q[0] = *(const v4u*)(ap);            // ds_load_b128
      f.q[1] = *(const v4u*)(ap + 16);       // ds_load_b128
      aF[mi] = f.v;
    }
#pragma unroll
    for (int ni = 0; ni < 2; ++ni) {         // B: cols n from transposed tile
      const __bf16* bp = Bs + (wn * 32 + ni * 16 + l15) * KL + kb;
      FragU f;
      f.q[0] = *(const v4u*)(bp);
      f.q[1] = *(const v4u*)(bp + 16);
      bF[ni] = f.v;
    }
#pragma unroll
    for (int mi = 0; mi < 2; ++mi)
#pragma unroll
      for (int ni = 0; ni < 2; ++ni)
        acc[mi][ni] = __builtin_amdgcn_wmma_f32_16x16x32_bf16(
            false, aF[mi], false, bF[ni], (short)0, acc[mi][ni], false, false);
  }

  // ---- Phase 4: epilogue  out = (acc + b[col]) * sqrt(2048) ----
  const float scale = 45.254833995939045f;   // sqrt(2048)
#pragma unroll
  for (int ni = 0; ni < 2; ++ni) {
    const int col = n0 + wn * 32 + ni * 16 + l15;
    const float bv = bias[col];
#pragma unroll
    for (int mi = 0; mi < 2; ++mi) {
      const int rb = m0 + wm * 32 + mi * 16 + half * 8;
#pragma unroll
      for (int v = 0; v < 8; ++v) {
        out[(long long)(rb + v) * NTOTAL + col] = (acc[mi][ni][v] + bv) * scale;
      }
    }
  }
}

// ---------------- host launch ----------------
extern "C" void kernel_launch(void* const* d_in, const int* in_sizes, int n_in,
                              void* d_out, int out_size, void* d_ws, size_t ws_size,
                              hipStream_t stream) {
  (void)in_sizes; (void)n_in; (void)out_size; (void)d_ws; (void)ws_size;

  const int*   x   = (const int*)  d_in[0];   // [4, 8192] int32 tokens
  const float* emb = (const float*)d_in[1];   // [50000, 300] f32
  const float* W   = (const float*)d_in[2];   // [300, 2048] f32
  const float* b   = (const float*)d_in[3];   // [2048] f32
  float*       out = (float*)d_out;           // [4, 8192, 2048] f32

  const size_t shmem = (size_t)(BM + BN) * KL * 2 + BM * sizeof(int);
  hipError_t err = hipFuncSetAttribute((const void*)emb_linear_wmma,
                      hipFuncAttributeMaxDynamicSharedMemorySize, (int)shmem);
  (void)err;

  dim3 grid(NTOTAL / BN, MTOTAL / BM);        // (32, 256)
  emb_linear_wmma<<<grid, dim3(256, 1, 1), (unsigned)shmem, stream>>>(x, emb, W, b, out);
}
